// HeteroSAGE_16398185136729
// MI455X (gfx1250) — compile-verified, matrix operand order checked by reading, not run on hardware
//
#include <hip/hip_runtime.h>

#define FD 128

typedef __attribute__((ext_vector_type(16))) __bf16 v16bf;
typedef __attribute__((ext_vector_type(8)))  float  v8f;

union Frag16 { unsigned u[8]; v16bf v; };

// Single-instruction packed f32->bf16 (RNE). Builtin not declared for gfx1250
// (probe-confirmed via __has_builtin), so emit the VOP3 directly.
__device__ __forceinline__ unsigned pack_bf16(float lo, float hi) {
    unsigned r;
    asm("v_cvt_pk_bf16_f32 %0, %1, %2" : "=v"(r) : "v"(lo), "v"(hi));
    return r;
}

__global__ __launch_bounds__(256)
void zero_f32(float* __restrict__ p, int n) {
    int i = blockIdx.x * 256 + threadIdx.x;
    if (i < n) p[i] = 0.0f;
}

// One edge per 32 lanes: coalesced float4 gather of source row, atomic scatter to dest.
__global__ __launch_bounds__(256)
void edge_scatter(const float* __restrict__ xs, const int* __restrict__ src,
                  const int* __restrict__ dst, float* __restrict__ agg,
                  float* __restrict__ cnt, int nE) {
    long long g = (long long)blockIdx.x * 256 + threadIdx.x;
    int e    = (int)(g >> 5);
    int lane = (int)(g & 31);
    if (e >= nE) return;
    int s = src[e], d = dst[e];
    float4 v = ((const float4*)(xs + (long long)s * FD))[lane];
    float* ap = agg + (long long)d * FD + lane * 4;
    atomicAdd(ap + 0, v.x);
    atomicAdd(ap + 1, v.y);
    atomicAdd(ap + 2, v.z);
    atomicAdd(ap + 3, v.w);
    if (lane == 0) atomicAdd(cnt + d, 1.0f);
}

// y[n,:] = (agg[n]/max(cnt,1)) @ Wl + x[n,:] @ Wr + bias  ; K=256 fused GEMM via WMMA bf16.
// Block = 256 threads = 8 waves; wave w computes rows [blk*128 + w*16, +16), all 128 cols.
__global__ __launch_bounds__(256)
void sage_gemm(const float* __restrict__ agg, const float* __restrict__ cnt,
               const float* __restrict__ x,
               const float* __restrict__ Wlw, const float* __restrict__ Wrw,
               const float* __restrict__ bias, float* __restrict__ y, int nrows) {
    // LDS: W packed as dwords holding bf16 pairs (k, k+1) for even k:
    //   Wp[(k>>1)*128 + n] = {bf16 W[k][n], bf16 W[k+1][n]}   (k in 0..255, n in 0..127)
    __shared__ unsigned Wp[128 * 128];   // 64 KB

    int tid = threadIdx.x;
    for (int d = tid; d < 128 * 128; d += 256) {
        int n  = d & 127;
        int k  = (d >> 7) * 2;
        float lo, hi;
        if (k < 128) { lo = Wlw[k * 128 + n];         hi = Wlw[(k + 1) * 128 + n]; }
        else         { lo = Wrw[(k - 128) * 128 + n]; hi = Wrw[(k - 127) * 128 + n]; }
        Wp[d] = pack_bf16(lo, hi);
    }
    __syncthreads();

    int wave  = tid >> 5;
    int lane  = tid & 31;
    int row16 = lane & 15;     // A row / B col / D col (per WMMA layout)
    int half  = lane >> 4;     // K-half selector
    int rowbase = blockIdx.x * 128 + wave * 16;
    int row  = rowbase + row16;
    int rowc = row < nrows ? row : nrows - 1;    // clamp loads; keep EXEC all-ones for WMMA

    float rs = 1.0f / fmaxf(cnt[rowc], 1.0f);
    const float2* aggr = (const float2*)(agg + (long long)rowc * FD);
    const float2* xr   = (const float2*)(x   + (long long)rowc * FD);

    v8f zero8 = {0.f, 0.f, 0.f, 0.f, 0.f, 0.f, 0.f, 0.f};
    v8f acc[8];
#pragma unroll
    for (int i = 0; i < 8; i++) acc[i] = zero8;

#pragma unroll
    for (int ks = 0; ks < 8; ks++) {
        // ---- A fragment: 16x32 bf16, row = row16, K = ks*32 .. ks*32+31 ----
        Frag16 a;
        int kbase = ks * 32 + half * 8;
#pragma unroll
        for (int j = 0; j < 8; j++) {
            int kk = kbase + (j & 3) * 2 + (j >> 2) * 16;   // even
            float2 p;
            if (kk < 128) { p = aggr[kk >> 1]; p.x *= rs; p.y *= rs; }
            else          { p = xr[(kk - 128) >> 1]; }
            a.u[j] = pack_bf16(p.x, p.y);
        }
        int kb2 = ks * 16 + half * 4;   // packed-dword row base
#pragma unroll
        for (int nt = 0; nt < 8; nt++) {
            // ---- B fragment: 32x16 bf16, col = row16, same K block ----
            Frag16 b;
            int n = nt * 16 + row16;
#pragma unroll
            for (int j = 0; j < 8; j++) {
                int k2i = kb2 + (j & 3) + (j >> 2) * 8;
                b.u[j] = Wp[k2i * 128 + n];
            }
            acc[nt] = __builtin_amdgcn_wmma_f32_16x16x32_bf16(
                false, a.v, false, b.v, (short)0, acc[nt], false, false);
        }
    }

    // ---- store: lane<16 -> (M=r, N=lane); lane>=16 -> (M=r+8, N=lane-16) ----
#pragma unroll
    for (int nt = 0; nt < 8; nt++) {
        int col = nt * 16 + row16;
        float bv = bias[col];
#pragma unroll
        for (int r = 0; r < 8; r++) {
            int orow = rowbase + r + half * 8;
            if (orow < nrows) y[(long long)orow * FD + col] = acc[nt][r] + bv;
        }
    }
}

// In-place LayerNorm(D=128) * gamma + beta, then ReLU. One wave (32 lanes) per row.
__global__ __launch_bounds__(256)
void ln_relu(float* __restrict__ xb, const float* __restrict__ g,
             const float* __restrict__ bt, int nrows) {
    int wave = threadIdx.x >> 5;
    int lane = threadIdx.x & 31;
    int row  = blockIdx.x * 8 + wave;
    if (row >= nrows) return;
    float4* rp = (float4*)(xb + (long long)row * FD);
    float4 v = rp[lane];

    float s = v.x + v.y + v.z + v.w;
#pragma unroll
    for (int m = 16; m > 0; m >>= 1) s += __shfl_xor(s, m, 32);
    float mu = s * (1.0f / 128.0f);

    float dx = v.x - mu, dy = v.y - mu, dz = v.z - mu, dw = v.w - mu;
    float q = dx * dx + dy * dy + dz * dz + dw * dw;
#pragma unroll
    for (int m = 16; m > 0; m >>= 1) q += __shfl_xor(q, m, 32);
    float inv = rsqrtf(q * (1.0f / 128.0f) + 1e-5f);

    float4 gg = ((const float4*)g)[lane];
    float4 bb = ((const float4*)bt)[lane];
    float4 o;
    o.x = fmaxf(dx * inv * gg.x + bb.x, 0.0f);
    o.y = fmaxf(dy * inv * gg.y + bb.y, 0.0f);
    o.z = fmaxf(dz * inv * gg.z + bb.z, 0.0f);
    o.w = fmaxf(dw * inv * gg.w + bb.w, 0.0f);
    rp[lane] = o;
}

extern "C" void kernel_launch(void* const* d_in, const int* in_sizes, int n_in,
                              void* d_out, int out_size, void* d_ws, size_t ws_size,
                              hipStream_t stream) {
    const float* x_user = (const float*)d_in[0];
    const float* x_item = (const float*)d_in[1];
    const int* u2i_src  = (const int*)d_in[2];
    const int* u2i_dst  = (const int*)d_in[3];
    const int* i2u_src  = (const int*)d_in[4];
    const int* i2u_dst  = (const int*)d_in[5];
    const float* Wl     = (const float*)d_in[6];   // [L,2,128,128]
    const float* Wr     = (const float*)d_in[7];   // [L,2,128,128]
    const float* bl     = (const float*)d_in[8];   // [L,2,128]
    const float* gamma  = (const float*)d_in[9];   // [L,2,128]
    const float* beta   = (const float*)d_in[10];  // [L,2,128]

    const int NU = in_sizes[0] / FD;
    const int NI = in_sizes[1] / FD;
    const int E  = in_sizes[2];

    float* ws   = (float*)d_ws;
    float* xu_a = ws;                               // [NU,128] layer-1 user out
    float* xi_a = xu_a + (size_t)NU * FD;           // [NI,128] layer-1 item out
    float* aggb = xi_a + (size_t)NI * FD;           // [NU,128] scatter accumulator (max size)
    float* cntb = aggb + (size_t)NU * FD;           // [NU]

    float* outU = (float*)d_out;                    // tuple order: (xu, xi)
    float* outI = outU + (size_t)NU * FD;

    const dim3 blk(256);
    const int scatterBlocks = (E * 32 + 255) / 256;

    for (int l = 0; l < 2; ++l) {
        const float* srcU = l ? xu_a : x_user;
        const float* srcI = l ? xi_a : x_item;
        float* dstU = (l == 1) ? outU : xu_a;
        float* dstI = (l == 1) ? outI : xi_a;

        // ---- direction 0: user -> item (W/b index l*2+0, LN index l*2+1) ----
        zero_f32<<<(NI * FD + 255) / 256, blk, 0, stream>>>(aggb, NI * FD);
        zero_f32<<<(NI + 255) / 256, blk, 0, stream>>>(cntb, NI);
        edge_scatter<<<scatterBlocks, blk, 0, stream>>>(srcU, u2i_src, u2i_dst, aggb, cntb, E);
        sage_gemm<<<(NI + 127) / 128, blk, 0, stream>>>(
            aggb, cntb, srcI,
            Wl + (size_t)(l * 2 + 0) * FD * FD, Wr + (size_t)(l * 2 + 0) * FD * FD,
            bl + (l * 2 + 0) * FD, dstI, NI);
        ln_relu<<<(NI + 7) / 8, blk, 0, stream>>>(dstI, gamma + (l * 2 + 1) * FD,
                                                  beta + (l * 2 + 1) * FD, NI);

        // ---- direction 1: item -> user (W/b index l*2+1, LN index l*2+0) ----
        zero_f32<<<(NU * FD + 255) / 256, blk, 0, stream>>>(aggb, NU * FD);
        zero_f32<<<(NU + 255) / 256, blk, 0, stream>>>(cntb, NU);
        edge_scatter<<<scatterBlocks, blk, 0, stream>>>(srcI, i2u_src, i2u_dst, aggb, cntb, E);
        sage_gemm<<<(NU + 127) / 128, blk, 0, stream>>>(
            aggb, cntb, srcU,
            Wl + (size_t)(l * 2 + 1) * FD * FD, Wr + (size_t)(l * 2 + 1) * FD * FD,
            bl + (l * 2 + 1) * FD, dstU, NU);
        ln_relu<<<(NU + 7) / 8, blk, 0, stream>>>(dstU, gamma + (l * 2 + 0) * FD,
                                                  beta + (l * 2 + 0) * FD, NU);
    }
}
